// HybridQLSTM_65481071395995
// MI455X (gfx1250) — compile-verified, matrix operand order checked by reading, not run on hardware
//
#include <hip/hip_runtime.h>
#include <hip/hip_bf16.h>
#include <stdint.h>

// ---------------------------------------------------------------------------
// Problem constants (from reference)
// ---------------------------------------------------------------------------
#define TT   2048
#define BB   64
#define DIN  256
#define DH   256
#define NG   1024          // 4 gates * 256
#define TAGS 64

typedef __attribute__((ext_vector_type(16))) _Float16 v16h;
typedef __attribute__((ext_vector_type(8)))  float    v8f;
typedef __attribute__((ext_vector_type(4)))  int      v4i;

union HF { v16h h; v4i i[2]; };

// A fragment (16x32 f16): lane L holds row M = L&15; halves 0..7 = K kb..kb+7,
// halves 8..15 = K 16+kb..16+kb+7, kb = (L>=16) ? 8 : 0.  Row data is
// contiguous in K, so the fragment is two 16B vector loads.
__device__ __forceinline__ v16h ld_a_frag(const _Float16* row, int lane) {
  int kb = (lane & 16) >> 1;               // 0 or 8 halves
  HF u;
  u.i[0] = *(const v4i*)(row + kb);
  u.i[1] = *(const v4i*)(row + 16 + kb);
  return u.h;
}

// B fragment from pre-packed weights: each lane's 16 halves are contiguous.
__device__ __forceinline__ v16h ld_b_frag(const _Float16* p) {
  HF u;
  u.i[0] = *(const v4i*)(p);
  u.i[1] = *(const v4i*)(p + 8);
  return u.h;
}

__device__ __forceinline__ v8f wmma16(v16h a, v16h b, v8f c) {
  return __builtin_amdgcn_wmma_f32_16x16x32_f16(false, a, false, b,
                                                (short)0, c, false, false);
}

__device__ __forceinline__ float sigm(float x) {
  return 1.0f / (1.0f + __expf(-x));
}
__device__ __forceinline__ float tanh_f(float x) {
  float e = __expf(-2.0f * x);
  return (1.0f - e) / (1.0f + e);
}

// ---------------------------------------------------------------------------
// Workspace layout (bytes, 256-aligned)
// ---------------------------------------------------------------------------
static constexpr size_t OFF_WXP  = 0;                         // 8*64*32*16 f16 = 512KB
static constexpr size_t SZ_GPACK = (size_t)8 * 64 * 32 * 16 * 2;
static constexpr size_t OFF_WHP  = OFF_WXP + SZ_GPACK;        // 512KB
static constexpr size_t OFF_WTP  = OFF_WHP + SZ_GPACK;        // 8*4*32*16 f16 = 32KB
static constexpr size_t OFF_BCAT = OFF_WTP + (size_t)8 * 4 * 32 * 16 * 2;
static constexpr size_t OFF_BTC  = OFF_BCAT + NG * 4;
static constexpr size_t OFF_XEMB = OFF_BTC + 256;             // T*B*256 f16 = 64MB
static constexpr size_t OFF_HOUT = OFF_XEMB + (size_t)TT * BB * DIN * 2;  // 64MB
static constexpr size_t OFF_XG   = OFF_HOUT + (size_t)TT * BB * DH * 2;   // 256MB
static constexpr size_t TOTAL_PRE = OFF_XG + (size_t)TT * BB * NG * 2;

// ---------------------------------------------------------------------------
// Kernel: pack the 4 gate weight matrices [512,256] f32 into WMMA-B-fragment
// f16 layout, split into x-part (rows 0..255) and h-part (rows 256..511),
// gate-concatenated to N=1024.  Packed index = ((kt*64+nt)*32 + lane)*16 + i.
// ---------------------------------------------------------------------------
__global__ __launch_bounds__(256) void pack_gate_weights(
    const float* Wf, const float* Wi, const float* Wg_, const float* Wo,
    _Float16* WxP, _Float16* WhP) {
  int idx = blockIdx.x * blockDim.x + threadIdx.x;
  if (idx >= 8 * 64 * 32 * 16) return;
  int i    = idx & 15;
  int lane = (idx >> 4) & 31;
  int nt   = (idx >> 9) & 63;
  int kt   = idx >> 15;
  int kb   = (lane & 16) ? 8 : 0;
  int kl   = (i < 8) ? (kb + i) : (16 + kb + (i - 8));
  int k    = kt * 32 + kl;
  int n    = nt * 16 + (lane & 15);
  int gate = n >> 8;
  int col  = n & 255;
  const float* W = (gate == 0) ? Wf : (gate == 1) ? Wi : (gate == 2) ? Wg_ : Wo;
  WxP[idx] = (_Float16)W[(size_t)k * DH + col];
  WhP[idx] = (_Float16)W[(size_t)(DIN + k) * DH + col];
}

// Pack tag weights [256,64] + biases.
__global__ __launch_bounds__(256) void pack_wt_bias(
    const float* Wt, const float* bf, const float* bi, const float* bg,
    const float* bo, const float* bt, _Float16* WtP, float* bcat, float* btc) {
  int idx = blockIdx.x * blockDim.x + threadIdx.x;
  if (idx < 8 * 4 * 32 * 16) {
    int i = idx & 15, lane = (idx >> 4) & 31, nt = (idx >> 9) & 3, kt = idx >> 11;
    int kb = (lane & 16) ? 8 : 0;
    int kl = (i < 8) ? (kb + i) : (16 + kb + (i - 8));
    int k = kt * 32 + kl, n = nt * 16 + (lane & 15);
    WtP[idx] = (_Float16)Wt[(size_t)k * TAGS + n];
  }
  if (idx < NG) {
    int gate = idx >> 8, j = idx & 255;
    const float* bp = (gate == 0) ? bf : (gate == 1) ? bi : (gate == 2) ? bg : bo;
    bcat[idx] = bp[j];
  }
  if (idx < TAGS) btc[idx] = bt[idx];
}

// Embedding gather -> f16 [T,B,256]
__global__ __launch_bounds__(256) void embed_kernel(
    const int* sentence, const float* emb, _Float16* xemb) {
  size_t e = (size_t)blockIdx.x * blockDim.x + threadIdx.x;
  if (e >= (size_t)TT * BB * DIN) return;
  int d = (int)(e & 255);
  size_t tb = e >> 8;
  int row = sentence[tb];
  xemb[e] = (_Float16)emb[(size_t)row * DIN + d];
}

// ---------------------------------------------------------------------------
// Kernel: precompute input-to-hidden pre-activations for ALL timesteps in
// parallel: xg[t] = x_t @ Wx + bias   (fully data-parallel, no recurrence).
// 1 block per t, 8 waves, x_t staged in LDS.
// ---------------------------------------------------------------------------
__global__ __launch_bounds__(256) void xgate_kernel(
    const _Float16* xemb, const _Float16* WxP, const float* bcat,
    _Float16* xgates) {
  __shared__ __align__(16) _Float16 xs[BB * DIN];   // 32KB
  int t = blockIdx.x;
  const v4i* src = (const v4i*)(xemb + (size_t)t * BB * DIN);
  v4i* dst = (v4i*)xs;
#pragma unroll
  for (int j = 0; j < 8; ++j) dst[threadIdx.x + j * 256] = src[threadIdx.x + j * 256];
  __syncthreads();

  int lane = threadIdx.x & 31, w = threadIdx.x >> 5;
  int ncol = lane & 15;
  for (int it = 0; it < 32; ++it) {
    int tile = w * 32 + it;
    int mt = tile >> 6, nt = tile & 63;
    float bias = bcat[nt * 16 + ncol];
    v8f acc;
#pragma unroll
    for (int r = 0; r < 8; ++r) acc[r] = bias;
#pragma unroll
    for (int kt = 0; kt < 8; ++kt) {
      v16h a = ld_a_frag(&xs[(mt * 16 + ncol) * DIN + kt * 32], lane);
      v16h b = ld_b_frag(WxP + ((size_t)(kt * 64 + nt) * 32 + lane) * 16);
      acc = wmma16(a, b, acc);
    }
#pragma unroll
    for (int r = 0; r < 8; ++r) {
      int m = mt * 16 + r + ((lane >> 4) << 3);
      int n = nt * 16 + ncol;
      xgates[((size_t)t * BB + m) * NG + n] = (_Float16)acc[r];
    }
  }
}

// ---------------------------------------------------------------------------
// Kernel: the serial LSTM recurrence.  ONE workgroup, 32 waves (1024 threads).
// h in LDS (f16), c held persistently in registers.  Each wave owns one
// 16-row batch tile (mt = w>>3) and two 16-col hidden tiles (jt0, jt0+1),
// and computes ALL FOUR gate tiles for them -> activations stay in-register.
// Per step: 2048 v_wmma_f32_16x16x32_f16 across the workgroup.
// ---------------------------------------------------------------------------
__global__ __launch_bounds__(1024) void lstm_kernel(
    const _Float16* xemb, const _Float16* xgates, const _Float16* WxP,
    const _Float16* WhP, const float* bcat, _Float16* hout, int use_pre) {
  __shared__ __align__(16) _Float16 hbuf[BB * DH];  // 32KB
  int tid = threadIdx.x, lane = tid & 31, w = tid >> 5;
  int mt = w >> 3;               // batch tile (0..3), shared by both groups
  int jt0 = (2 * w) & 15;        // first hidden tile; second is jt0+1
  int ncol = lane & 15;
  int rofs = (lane >> 4) << 3;   // +8 rows for upper half-wave in C/D frags

  // zero h
  v4i z = {0, 0, 0, 0};
  v4i* hb = (v4i*)hbuf;
  hb[tid] = z;
  hb[tid + 1024] = z;

  float bsv[2][4];
#pragma unroll
  for (int q = 0; q < 2; ++q)
#pragma unroll
    for (int g = 0; g < 4; ++g)
      bsv[q][g] = bcat[g * 256 + (jt0 + q) * 16 + ncol];

  v8f cst[2];
#pragma unroll
  for (int q = 0; q < 2; ++q)
#pragma unroll
    for (int r = 0; r < 8; ++r) cst[q][r] = 0.0f;

  __syncthreads();

  for (int t = 0; t < TT; ++t) {
    v8f acc[2][4];
    if (use_pre) {
      // accumulator := bias + precomputed x-part pre-activation
#pragma unroll
      for (int q = 0; q < 2; ++q)
#pragma unroll
        for (int g = 0; g < 4; ++g) {
          int n = g * 256 + (jt0 + q) * 16 + ncol;
#pragma unroll
          for (int r = 0; r < 8; ++r) {
            int m = mt * 16 + r + rofs;
            acc[q][g][r] = bsv[q][g] +
                (float)xgates[((size_t)t * BB + m) * NG + n];
          }
        }
    } else {
      // fallback: bias init + x GEMM fused here (K = 256 over x)
#pragma unroll
      for (int q = 0; q < 2; ++q)
#pragma unroll
        for (int g = 0; g < 4; ++g)
#pragma unroll
          for (int r = 0; r < 8; ++r) acc[q][g][r] = bsv[q][g];
#pragma unroll
      for (int kt = 0; kt < 8; ++kt) {
        const _Float16* arow =
            xemb + ((size_t)t * BB + mt * 16 + ncol) * DIN + kt * 32;
        v16h a = ld_a_frag(arow, lane);
#pragma unroll
        for (int q = 0; q < 2; ++q)
#pragma unroll
          for (int g = 0; g < 4; ++g) {
            v16h b = ld_b_frag(
                WxP + ((size_t)(kt * 64 + g * 16 + jt0 + q) * 32 + lane) * 16);
            acc[q][g] = wmma16(a, b, acc[q][g]);
          }
      }
    }

    // hidden-to-hidden GEMM: K = 256 over previous h (LDS)
#pragma unroll
    for (int kt = 0; kt < 8; ++kt) {
      v16h a = ld_a_frag(&hbuf[(mt * 16 + ncol) * DH + kt * 32], lane);
#pragma unroll
      for (int q = 0; q < 2; ++q)
#pragma unroll
        for (int g = 0; g < 4; ++g) {
          v16h b = ld_b_frag(
              WhP + ((size_t)(kt * 64 + g * 16 + jt0 + q) * 32 + lane) * 16);
          acc[q][g] = wmma16(a, b, acc[q][g]);
        }
    }

    __syncthreads();  // all waves done reading hbuf for step t

    // in-register activations + cell update; write new h
#pragma unroll
    for (int q = 0; q < 2; ++q) {
      int jcol = (jt0 + q) * 16 + ncol;
#pragma unroll
      for (int r = 0; r < 8; ++r) {
        int m = mt * 16 + r + rofs;
        float fg = sigm(acc[q][0][r]);
        float ig = sigm(acc[q][1][r]);
        float gg = tanh_f(acc[q][2][r]);
        float og = sigm(acc[q][3][r]);
        float c = fg * cst[q][r] + ig * gg;
        cst[q][r] = c;
        float h = og * tanh_f(c);
        _Float16 hh = (_Float16)h;
        hbuf[m * DH + jcol] = hh;
        hout[((size_t)t * BB + m) * DH + jcol] = hh;
      }
    }
    __syncthreads();  // new h visible before next step's reads
  }
}

// ---------------------------------------------------------------------------
// Kernel: tag projection + log_softmax over the BATCH axis (reference uses
// axis=1).  One block per t: logits[64,64] = h_t @ Wt + bt via WMMA, then
// column-wise (over b) log-sum-exp.
// ---------------------------------------------------------------------------
__global__ __launch_bounds__(256) void tag_kernel(
    const _Float16* hout, const _Float16* WtP, const float* btc, float* out) {
  __shared__ __align__(16) _Float16 hs[BB * DH];  // 32KB
  __shared__ float lg[BB * TAGS];                 // 16KB
  int t = blockIdx.x;
  const v4i* src = (const v4i*)(hout + (size_t)t * BB * DH);
  v4i* dst = (v4i*)hs;
#pragma unroll
  for (int j = 0; j < 8; ++j) dst[threadIdx.x + j * 256] = src[threadIdx.x + j * 256];
  __syncthreads();

  int lane = threadIdx.x & 31, w = threadIdx.x >> 5;
  int ncol = lane & 15;
#pragma unroll
  for (int q = 0; q < 2; ++q) {
    int tile = w * 2 + q;
    int mt = tile >> 2, nt = tile & 3;
    float bias = btc[nt * 16 + ncol];
    v8f acc;
#pragma unroll
    for (int r = 0; r < 8; ++r) acc[r] = bias;
#pragma unroll
    for (int kt = 0; kt < 8; ++kt) {
      v16h a = ld_a_frag(&hs[(mt * 16 + ncol) * DH + kt * 32], lane);
      v16h b = ld_b_frag(WtP + ((size_t)(kt * 4 + nt) * 32 + lane) * 16);
      acc = wmma16(a, b, acc);
    }
#pragma unroll
    for (int r = 0; r < 8; ++r) {
      int m = mt * 16 + r + ((lane >> 4) << 3);
      int n = nt * 16 + ncol;
      lg[m * TAGS + n] = acc[r];
    }
  }
  __syncthreads();

  if (threadIdx.x < TAGS) {
    int j = threadIdx.x;
    float mx = -1e30f;
    for (int b = 0; b < BB; ++b) mx = fmaxf(mx, lg[b * TAGS + j]);
    float s = 0.0f;
    for (int b = 0; b < BB; ++b) s += __expf(lg[b * TAGS + j] - mx);
    float ls = mx + __logf(s);
    for (int b = 0; b < BB; ++b)
      out[(size_t)t * (BB * TAGS) + (size_t)b * TAGS + j] = lg[b * TAGS + j] - ls;
  }
}

// ---------------------------------------------------------------------------
extern "C" void kernel_launch(void* const* d_in, const int* in_sizes, int n_in,
                              void* d_out, int out_size, void* d_ws,
                              size_t ws_size, hipStream_t stream) {
  const int*   sentence = (const int*)d_in[0];
  const float* emb = (const float*)d_in[1];
  const float* Wf  = (const float*)d_in[2];
  const float* bf  = (const float*)d_in[3];
  const float* Wi  = (const float*)d_in[4];
  const float* bi  = (const float*)d_in[5];
  const float* Wg  = (const float*)d_in[6];
  const float* bg  = (const float*)d_in[7];
  const float* Wo  = (const float*)d_in[8];
  const float* bo  = (const float*)d_in[9];
  const float* Wt  = (const float*)d_in[10];
  const float* bt  = (const float*)d_in[11];
  float* out = (float*)d_out;

  char* ws = (char*)d_ws;
  _Float16* WxP   = (_Float16*)(ws + OFF_WXP);
  _Float16* WhP   = (_Float16*)(ws + OFF_WHP);
  _Float16* WtP   = (_Float16*)(ws + OFF_WTP);
  float*    bcat  = (float*)(ws + OFF_BCAT);
  float*    btc   = (float*)(ws + OFF_BTC);
  _Float16* xemb  = (_Float16*)(ws + OFF_XEMB);
  _Float16* hout  = (_Float16*)(ws + OFF_HOUT);
  _Float16* xgate = (_Float16*)(ws + OFF_XG);

  int use_pre = (ws_size >= TOTAL_PRE) ? 1 : 0;

  pack_gate_weights<<<(8 * 64 * 32 * 16) / 256, 256, 0, stream>>>(
      Wf, Wi, Wg, Wo, WxP, WhP);
  pack_wt_bias<<<64, 256, 0, stream>>>(Wt, bf, bi, bg, bo, bt, WtP, bcat, btc);
  embed_kernel<<<(TT * BB * DIN) / 256, 256, 0, stream>>>(sentence, emb, xemb);
  if (use_pre)
    xgate_kernel<<<TT, 256, 0, stream>>>(xemb, WxP, bcat, xgate);
  lstm_kernel<<<1, 1024, 0, stream>>>(xemb, xgate, WxP, WhP, bcat, hout,
                                      use_pre);
  tag_kernel<<<TT, 256, 0, stream>>>(hout, WtP, btc, out);
}
